// ScalableMoE_4681514352740
// MI455X (gfx1250) — matrix operand order repaired and checked by reading, not compiled, not run on hardware
//
#include <hip/hip_runtime.h>
#include <hip/hip_bf16.h>

// ---------------------------------------------------------------------------
// MoE top-2 FFN for MI455X (gfx1250, wave32, WMMA).
//   N=8192 tokens, H=1024, F=4096, E=8, top-2  => 275 GFLOP sparse (vs 1.1T dense)
// Pipeline: cvt x->bf16 | cvt+transpose w1,w2->bf16 | router | scan | scatter |
//           fused grouped WMMA FFN (gelu between GEMMs) | deterministic combine
// bf16 weights (134 MB total) are fully L2-resident (192 MB) -> the grouped
// GEMM streams weights at L2 bandwidth and runs at the bf16 WMMA roofline.
// ---------------------------------------------------------------------------

#define HIDDEN   1024
#define FFN_DIM  4096
#define NEXP     8
#define NTOK     8192
#define SLOTS    (NTOK * 2)
#define MBLK     64            // tokens per expert block
#define FCHUNK   128           // F processed per fused chunk
#define MAXTILES (SLOTS / MBLK + NEXP)

typedef __bf16 bf16_t;
typedef bf16_t v16bf __attribute__((ext_vector_type(16)));
typedef float  v8f   __attribute__((ext_vector_type(8)));

union FragBF { v16bf v; uint4 q[2]; unsigned short u[16]; };
union AccF   { v8f v; float f[8]; };

__device__ __forceinline__ unsigned short f32_to_bf16_rne(float f) {
  union { float f; unsigned u; } cv; cv.f = f;
  unsigned r = cv.u + 0x7FFFu + ((cv.u >> 16) & 1u);
  return (unsigned short)(r >> 16);
}

__device__ __forceinline__ float gelu_exact(float v) {
  return 0.5f * v * (1.0f + erff(v * 0.7071067811865475f));
}

// A fragment (16x32 bf16, M x K) from a row-major ushort buffer (row stride ld).
// ISA 16-bit A layout: lane r=lane&15 holds row r; g=lane>>4 selects K-halves:
//   elems 0-7  -> K = kbase + 8g + 0..7       (one b128)
//   elems 8-15 -> K = kbase + 16 + 8g + 0..7  (one b128)
__device__ __forceinline__ v16bf load_frag_a(const unsigned short* base, int ld,
                                             int kbase, int lane) {
  const int g = (lane >> 4) & 1;
  const int r = lane & 15;
  const unsigned short* p = base + (size_t)r * ld + kbase + g * 8;
  FragBF fr;
  fr.q[0] = *(const uint4*)(p);
  fr.q[1] = *(const uint4*)(p + 16);
  return fr.v;
}

// B fragment (32x16 bf16, K x N) from TRANSPOSED weights W[n][k] (row stride ldk).
// ISA 16-bit B layout: lane r=lane&15 holds column n=nbase+r;
//   elems 0-7 -> K = kbase + 8g + 0..7, elems 8-15 -> K = kbase + 16 + 8g + 0..7 (two b128 loads)
__device__ __forceinline__ v16bf load_frag_b(const unsigned short* W, int ldk,
                                             int nbase, int kbase, int lane) {
  const int g = (lane >> 4) & 1;
  const int r = lane & 15;
  const unsigned short* p = W + (size_t)(nbase + r) * ldk + kbase + g * 8;
  FragBF fr;
  fr.q[0] = *(const uint4*)(p);
  fr.q[1] = *(const uint4*)(p + 16);
  return fr.v;
}

__device__ __forceinline__ v8f wmma_bf16(v16bf a, v16bf b, v8f c) {
  return __builtin_amdgcn_wmma_f32_16x16x32_bf16(false, a, false, b,
                                                 (short)0, c, false, false);
}

// ---------------------------------------------------------------------------
__global__ void moe_init_kernel(int* counts) {
  if (threadIdx.x < NEXP) counts[threadIdx.x] = 0;
}

__global__ __launch_bounds__(256)
void moe_cvt_x_kernel(const float* __restrict__ x, unsigned short* __restrict__ x_bf) {
  const size_t i = ((size_t)blockIdx.x * blockDim.x + threadIdx.x) * 4;
  if (i >= (size_t)NTOK * HIDDEN) return;
  const float4 v = *(const float4*)(x + i);
  unsigned short o[4];
  o[0] = f32_to_bf16_rne(v.x); o[1] = f32_to_bf16_rne(v.y);
  o[2] = f32_to_bf16_rne(v.z); o[3] = f32_to_bf16_rne(v.w);
  *(uint2*)(x_bf + i) = *(uint2*)o;
}

// src: [E][rows][cols] fp32  ->  dst: [E][cols][rows] bf16   (tiled via LDS)
__global__ __launch_bounds__(256)
void moe_cvt_transpose_kernel(const float* __restrict__ src,
                              unsigned short* __restrict__ dst,
                              int rows, int cols) {
  __shared__ unsigned short tile[64][65];
  const int tiles_c = cols / 64;
  const int tiles_r = rows / 64;
  const int per_exp = tiles_c * tiles_r;
  const int e  = blockIdx.x / per_exp;
  const int t  = blockIdx.x % per_exp;
  const int r0 = (t / tiles_c) * 64;
  const int c0 = (t % tiles_c) * 64;
  const float* s = src + (size_t)e * rows * cols;
  unsigned short* d = dst + (size_t)e * rows * cols;
  for (int i = threadIdx.x; i < 64 * 64; i += 256) {
    const int r = i >> 6, c = i & 63;
    tile[c][r] = f32_to_bf16_rne(s[(size_t)(r0 + r) * cols + (c0 + c)]);
  }
  __syncthreads();
  for (int i = threadIdx.x; i < 64 * 64; i += 256) {
    const int c = i >> 6, r = i & 63;
    d[(size_t)(c0 + c) * rows + (r0 + r)] = tile[c][r];
  }
}

// One wave (32 lanes) per token: 8 dot products, shfl reduce, top-2 weights.
__global__ __launch_bounds__(256)
void moe_router_kernel(const float* __restrict__ x, const float* __restrict__ rw,
                       const float* __restrict__ rb, int* __restrict__ counts,
                       int* __restrict__ topk_idx, float* __restrict__ topk_wt) {
  const int lane = threadIdx.x & 31;
  const int wave = threadIdx.x >> 5;
  const int tok  = blockIdx.x * (blockDim.x >> 5) + wave;
  if (tok >= NTOK) return;
  const float* xr = x + (size_t)tok * HIDDEN;
  float acc[NEXP];
  #pragma unroll
  for (int e = 0; e < NEXP; ++e) acc[e] = 0.0f;
  for (int h = lane; h < HIDDEN; h += 32) {
    const float xv = xr[h];
    #pragma unroll
    for (int e = 0; e < NEXP; ++e) acc[e] += xv * rw[e * HIDDEN + h];
  }
  #pragma unroll
  for (int e = 0; e < NEXP; ++e)
    #pragma unroll
    for (int off = 16; off > 0; off >>= 1)
      acc[e] += __shfl_xor(acc[e], off, 32);
  if (lane == 0) {
    float lg[NEXP];
    #pragma unroll
    for (int e = 0; e < NEXP; ++e) lg[e] = acc[e] + rb[e];
    int i0 = 0;
    #pragma unroll
    for (int e = 1; e < NEXP; ++e) if (lg[e] > lg[i0]) i0 = e;      // lowest-idx wins ties
    int i1 = (i0 == 0) ? 1 : 0;
    #pragma unroll
    for (int e = 0; e < NEXP; ++e) if (e != i0 && lg[e] > lg[i1]) i1 = e;
    // softmax top-2 renormalized == 1/(1+exp(l1-l0))
    const float w0 = 1.0f / (1.0f + expf(lg[i1] - lg[i0]));
    topk_idx[tok * 2 + 0] = i0; topk_idx[tok * 2 + 1] = i1;
    topk_wt[tok * 2 + 0] = w0;  topk_wt[tok * 2 + 1] = 1.0f - w0;
    atomicAdd(&counts[i0], 1);
    atomicAdd(&counts[i1], 1);
  }
}

__global__ void moe_scan_kernel(const int* counts, int* offsets, int* tile_off,
                                int* cursors) {
  if (threadIdx.x == 0 && blockIdx.x == 0) {
    int o = 0, t = 0;
    for (int e = 0; e < NEXP; ++e) {
      offsets[e] = o; tile_off[e] = t; cursors[e] = o;
      o += counts[e];
      t += (counts[e] + MBLK - 1) / MBLK;
    }
    offsets[NEXP] = o; tile_off[NEXP] = t;
  }
}

__global__ __launch_bounds__(256)
void moe_scatter_kernel(const int* __restrict__ topk_idx,
                        const float* __restrict__ topk_wt, int* cursors,
                        int* perm_token, float* perm_wt, int* slot_of) {
  const int n = blockIdx.x * blockDim.x + threadIdx.x;
  if (n >= NTOK) return;
  #pragma unroll
  for (int k = 0; k < 2; ++k) {
    const int e = topk_idx[n * 2 + k];
    const int slot = atomicAdd(&cursors[e], 1);
    perm_token[slot] = n;
    perm_wt[slot]    = topk_wt[n * 2 + k];
    slot_of[n * 2 + k] = slot;
  }
}

// ---------------------------------------------------------------------------
// Fused grouped FFN: block = 64 tokens of one expert, 16 waves (512 threads).
// x tile resident in LDS (128 KB). For each F-chunk of 128:
//   phase1: each wave 2 h-tiles (K=1024 WMMA loop), gelu -> h_lds bf16
//   phase2: each wave 16 out-tiles (K=128 WMMA loop) into persistent v8f accs.
// Per-lane global_prefetch of next chunk's weight cachelines hides the cold
// L2->WGP$ latency across the chunk barriers. Combine weight folded into the
// store; padded rows masked out.
// ---------------------------------------------------------------------------
__global__ __launch_bounds__(512, 1)
void moe_expert_kernel(const unsigned short* __restrict__ x_bf,
                       const unsigned short* __restrict__ w1t,   // [E][F][H] bf16
                       const unsigned short* __restrict__ w2t,   // [E][H][F] bf16
                       const int*   __restrict__ perm_token,
                       const float* __restrict__ perm_wt,
                       const int*   __restrict__ offsets,
                       const int*   __restrict__ tile_off,
                       float* __restrict__ perm_out) {            // [SLOTS][H]
  __shared__ unsigned short x_lds[MBLK * HIDDEN];  // 128 KB
  __shared__ unsigned short h_lds[MBLK * FCHUNK];  //  16 KB

  const int b = blockIdx.x;
  if (b >= tile_off[NEXP]) return;
  int e = 0;
  while (e < NEXP - 1 && b >= tile_off[e + 1]) ++e;
  const int s0 = offsets[e] + (b - tile_off[e]) * MBLK;
  int cnt = offsets[e + 1] - s0;
  if (cnt > MBLK) cnt = MBLK;

  const int tid = threadIdx.x;
  for (int i = tid; i < MBLK * (HIDDEN / 8); i += blockDim.x) {
    const int row = i / (HIDDEN / 8);
    const int c   = (i % (HIDDEN / 8)) * 8;
    uint4 val = make_uint4(0u, 0u, 0u, 0u);
    if (row < cnt) {
      const int tok = perm_token[s0 + row];
      val = *(const uint4*)(x_bf + (size_t)tok * HIDDEN + c);
    }
    *(uint4*)(&x_lds[row * HIDDEN + c]) = val;
  }
  __syncthreads();

  const int wave = tid >> 5;
  const int lane = tid & 31;
  const int g    = (lane >> 4) & 1;
  const int lrow = lane & 15;
  const int mi   = wave & 3;   // M tile (both phases)
  const int f2   = wave >> 2;  // 0..3: phase1 f-pair group / phase2 256-col band

  const unsigned short* w1e = w1t + (size_t)e * FFN_DIM * HIDDEN;
  const unsigned short* w2e = w2t + (size_t)e * HIDDEN * FFN_DIM;

  const v8f vzero = {0.f, 0.f, 0.f, 0.f, 0.f, 0.f, 0.f, 0.f};
  v8f acc[16];
  #pragma unroll
  for (int j = 0; j < 16; ++j) acc[j] = vzero;

  for (int f0 = 0; f0 < FFN_DIM; f0 += FCHUNK) {
    // Prefetch next chunk's weight streams (emits global_prefetch_b8): each
    // lane touches the cacheline of the first B fragment it will need.
    const int f0n = f0 + FCHUNK;
    if (f0n < FFN_DIM) {
      __builtin_prefetch(w1e + (size_t)(f0n + f2 * 32 + lrow) * HIDDEN, 0, 3);
      __builtin_prefetch(w2e + (size_t)(f2 * 256 + lrow) * FFN_DIM + f0n, 0, 3);
    }
    // -------- phase 1: h[64][FCHUNK] = gelu(x @ w1[:, f0:f0+FCHUNK]) --------
    v8f c1_0 = vzero, c1_1 = vzero;
    const int fcol0 = f0 + (f2 * 2 + 0) * 16;
    const int fcol1 = f0 + (f2 * 2 + 1) * 16;
    for (int k = 0; k < HIDDEN; k += 32) {
      const v16bf a  = load_frag_a(&x_lds[mi * 16 * HIDDEN], HIDDEN, k, lane);
      const v16bf b0 = load_frag_b(w1e, HIDDEN, fcol0, k, lane);
      const v16bf b1 = load_frag_b(w1e, HIDDEN, fcol1, k, lane);
      c1_0 = wmma_bf16(a, b0, c1_0);
      c1_1 = wmma_bf16(a, b1, c1_1);
    }
    __syncthreads();  // previous chunk's phase-2 readers of h_lds are done
    AccF u0, u1; u0.v = c1_0; u1.v = c1_1;
    #pragma unroll
    for (int r = 0; r < 8; ++r) {
      const int m = mi * 16 + g * 8 + r;  // C/D layout: M = vgpr + 8*(lane/16)
      h_lds[m * FCHUNK + (f2 * 2 + 0) * 16 + lrow] =
          f32_to_bf16_rne(gelu_exact(u0.f[r]));
      h_lds[m * FCHUNK + (f2 * 2 + 1) * 16 + lrow] =
          f32_to_bf16_rne(gelu_exact(u1.f[r]));
    }
    __syncthreads();
    // -------- phase 2: acc += h @ w2[f0:f0+FCHUNK, :] --------
    for (int kk = 0; kk < FCHUNK; kk += 32) {
      const v16bf a2 = load_frag_a(&h_lds[mi * 16 * FCHUNK], FCHUNK, kk, lane);
      #pragma unroll
      for (int j = 0; j < 16; ++j) {
        const int ncol = f2 * 256 + j * 16;
        const v16bf b2 = load_frag_b(w2e, FFN_DIM, ncol, f0 + kk, lane);
        acc[j] = wmma_bf16(a2, b2, acc[j]);
      }
    }
  }

  float wsc[8];
  #pragma unroll
  for (int r = 0; r < 8; ++r) {
    const int m = mi * 16 + g * 8 + r;
    wsc[r] = (m < cnt) ? perm_wt[s0 + m] : 0.0f;
  }
  #pragma unroll
  for (int j = 0; j < 16; ++j) {
    AccF u; u.v = acc[j];
    const int ncol = f2 * 256 + j * 16 + lrow;
    #pragma unroll
    for (int r = 0; r < 8; ++r) {
      const int m = mi * 16 + g * 8 + r;
      if (m < cnt)
        perm_out[(size_t)(s0 + m) * HIDDEN + ncol] = u.f[r] * wsc[r];
    }
  }
}

__global__ __launch_bounds__(256)
void moe_combine_kernel(const float* __restrict__ perm_out,
                        const int* __restrict__ slot_of, float* __restrict__ out) {
  const size_t i = (size_t)blockIdx.x * blockDim.x + threadIdx.x;
  if (i >= (size_t)NTOK * HIDDEN) return;
  const int n = (int)(i >> 10);
  const int h = (int)(i & (HIDDEN - 1));
  const int sA = slot_of[n * 2 + 0], sB = slot_of[n * 2 + 1];
  out[i] = perm_out[(size_t)sA * HIDDEN + h] + perm_out[(size_t)sB * HIDDEN + h];
}

// ---------------------------------------------------------------------------
extern "C" void kernel_launch(void* const* d_in, const int* in_sizes, int n_in,
                              void* d_out, int out_size, void* d_ws, size_t ws_size,
                              hipStream_t stream) {
  const float* x  = (const float*)d_in[0];
  const float* rw = (const float*)d_in[1];
  const float* rb = (const float*)d_in[2];
  const float* w1 = (const float*)d_in[3];
  const float* w2 = (const float*)d_in[4];
  float* out = (float*)d_out;
  (void)in_sizes; (void)n_in; (void)out_size; (void)ws_size;

  char* ws = (char*)d_ws;
  size_t off = 0;
  auto carve = [&](size_t bytes) -> void* {
    void* p = (void*)(ws + off);
    off += (bytes + 255) & ~(size_t)255;
    return p;
  };
  int*   counts   = (int*)carve(sizeof(int) * NEXP);
  int*   offsets  = (int*)carve(sizeof(int) * (NEXP + 1));
  int*   tile_off = (int*)carve(sizeof(int) * (NEXP + 1));
  int*   cursors  = (int*)carve(sizeof(int) * NEXP);
  int*   topk_idx = (int*)carve(sizeof(int) * NTOK * 2);
  float* topk_wt  = (float*)carve(sizeof(float) * NTOK * 2);
  int*   slot_of  = (int*)carve(sizeof(int) * NTOK * 2);
  int*   perm_tok = (int*)carve(sizeof(int) * SLOTS);
  float* perm_wt  = (float*)carve(sizeof(float) * SLOTS);
  unsigned short* x_bf = (unsigned short*)carve(sizeof(unsigned short) * (size_t)NTOK * HIDDEN);
  unsigned short* w1t  = (unsigned short*)carve(sizeof(unsigned short) * (size_t)NEXP * HIDDEN * FFN_DIM);
  unsigned short* w2t  = (unsigned short*)carve(sizeof(unsigned short) * (size_t)NEXP * HIDDEN * FFN_DIM);
  float* perm_out = (float*)carve(sizeof(float) * (size_t)SLOTS * HIDDEN);

  moe_init_kernel<<<1, 32, 0, stream>>>(counts);
  moe_cvt_x_kernel<<<(NTOK * HIDDEN / 4 + 255) / 256, 256, 0, stream>>>(x, x_bf);
  moe_cvt_transpose_kernel<<<NEXP * (HIDDEN / 64) * (FFN_DIM / 64), 256, 0, stream>>>(
      w1, w1t, HIDDEN, FFN_DIM);
  moe_cvt_transpose_kernel<<<NEXP * (FFN_DIM / 64) * (HIDDEN / 64), 256, 0, stream>>>(
      w2, w2t, FFN_DIM, HIDDEN);
  moe_router_kernel<<<NTOK / 8, 256, 0, stream>>>(x, rw, rb, counts, topk_idx, topk_wt);
  moe_scan_kernel<<<1, 32, 0, stream>>>(counts, offsets, tile_off, cursors);
  moe_scatter_kernel<<<(NTOK + 255) / 256, 256, 0, stream>>>(
      topk_idx, topk_wt, cursors, perm_tok, perm_wt, slot_of);
  moe_expert_kernel<<<MAXTILES, 512, 0, stream>>>(
      x_bf, w1t, w2t, perm_tok, perm_wt, offsets, tile_off, perm_out);
  moe_combine_kernel<<<(NTOK * HIDDEN + 255) / 256, 256, 0, stream>>>(
      perm_out, slot_of, out);
}
